// SPSTPAttention_15891378995683
// MI455X (gfx1250) — matrix operand order repaired and checked by reading, not compiled, not run on hardware
//
#include <hip/hip_runtime.h>
#include <hip/hip_bf16.h>
#include <stdint.h>

// ---------------------------------------------------------------------------
// SPSTP attention for gfx1250 (MI455X), wave32 + v_wmma_f32_16x16x32_f16.
// Pipeline:
//   1) f32->f16 converts (x, Wq, Wk, Wv, Wo)
//   2) WMMA GEMM: Q = x@Wq^T+bq  -> Qa[b,h,t,0:64]   (row width 96)
//                 K = x@Wk^T+bk  -> Ka[b,h,t,0:64]
//                 V = x@Wv^T+bv  -> Vt[b,h,col,key]  (transposed)
//   3) STP kernel: Qa[...,64:67]=q@Wqs^T+bqs, Ka[...,64:67]=cross(k3,v3),
//                  tail 67..95 zeroed (padding for K=96 contraction)
//   4) flash attention per (b,h,16 query rows): scores = Qa@Ka^T (K=96,
//      3 wmma chunks) * temp, online softmax, O += P@V (P via LDS relayout)
//   5) WMMA GEMM: out = O@Wo^T + bo (f32 store)
// ---------------------------------------------------------------------------

#define BATCH 2
#define SEQ   2048
#define DIM   1024
#define HEADS 16
#define HDIM  64
#define AUGW  96        // 64 + 3 STP dims + 29 zero pad -> 3 x K32 wmma chunks
#define BHT   (BATCH*HEADS*SEQ)
#define MROWS (BATCH*SEQ)   // 4096

typedef __attribute__((ext_vector_type(16))) _Float16 v16h;
typedef __attribute__((ext_vector_type(8)))  float    v8f;

union FragU { v16h h; uint32_t u[8]; };

__device__ __forceinline__ v8f zero_v8f() {
    v8f z;
#pragma unroll
    for (int i = 0; i < 8; ++i) z[i] = 0.0f;
    return z;
}

__device__ __forceinline__ v8f wmma16(v16h a, v16h b, v8f c) {
    // D = A(16x32 f16) * B(32x16 f16) + C(16x16 f32)
    return __builtin_amdgcn_wmma_f32_16x16x32_f16(
        false, a, false, b, (short)0, c, false, false);
}

// A-fragment: 16x32 f16, A row-major, lda elements. ISA 7.12.2 layout:
// lane L: m=L&15, h=L>>4; VGPR i<4: K=8h+2i(+1); VGPR i>=4: K=16+8h+2(i-4).
__device__ __forceinline__ v16h load_a_frag(const _Float16* A, int lda,
                                            int m0, int kbase, int lane) {
    const int m = lane & 15, hh = lane >> 4;
    const _Float16* row = A + (size_t)(m0 + m) * lda;
    FragU f;
#pragma unroll
    for (int i = 0; i < 4; ++i)
        f.u[i] = *(const uint32_t*)(row + kbase + 8 * hh + 2 * i);
#pragma unroll
    for (int i = 0; i < 4; ++i)
        f.u[4 + i] = *(const uint32_t*)(row + kbase + 16 + 8 * hh + 2 * i);
    return f.h;
}

// B-fragment: 32x16 f16 where element (k,n) = Bt[(n0+n)*ldb + kbase + k]
// (contraction along the fast dim of Bt). Layout: lane L: n=L&15, h=L>>4;
// VGPR i<4: K=8h+2i(+1); VGPR i>=4: K=16+8h+2(i-4)(+1).
__device__ __forceinline__ v16h load_b_frag(const _Float16* Bt, int ldb,
                                            int n0, int kbase, int lane) {
    const int n = lane & 15, hh = lane >> 4;
    const _Float16* row = Bt + (size_t)(n0 + n) * ldb;
    FragU f;
#pragma unroll
    for (int i = 0; i < 8; ++i)
        f.u[i] = *(const uint32_t*)(row + kbase + (i < 4 ? 8 * hh + 2 * i : 16 + 8 * hh + 2 * (i - 4)));
    return f.h;
}

// ---------------------------------------------------------------------------
__global__ void __launch_bounds__(256)
cvt_f32_to_f16(const float* __restrict__ in, _Float16* __restrict__ out, int n) {
    int i = blockIdx.x * 256 + threadIdx.x;
    if (i < n) out[i] = (_Float16)in[i];
}

// ---------------------------------------------------------------------------
// 64x64-per-wave WMMA GEMM: C[m,n] = sum_k A[m,k]*Bt[n,k] + bias[n]
// mode 0: f16 store split-head  -> out[((b*H+h)*T+t)*96 + hd]
// mode 1: f16 store transposed  -> out[((b*H+h)*64+hd)*T + t]
// mode 2: f32 store row-major   -> out[m*N + n]
__global__ void __launch_bounds__(32)
gemm_wmma_64x64(const _Float16* __restrict__ A, const _Float16* __restrict__ Bt,
                const float* __restrict__ bias, void* __restrict__ outp,
                int M, int N, int K, int mode) {
    const int lane   = threadIdx.x;
    const int tilesN = N >> 6;
    const int bm     = blockIdx.x / tilesN;
    const int bn     = blockIdx.x % tilesN;
    const int m0     = bm * 64, n0 = bn * 64;

    v8f acc[4][4];
#pragma unroll
    for (int i = 0; i < 4; ++i)
#pragma unroll
        for (int j = 0; j < 4; ++j) acc[i][j] = zero_v8f();

    for (int k = 0; k < K; k += 32) {
        v16h a[4], b[4];
#pragma unroll
        for (int i = 0; i < 4; ++i) a[i] = load_a_frag(A, K, m0 + 16 * i, k, lane);
#pragma unroll
        for (int j = 0; j < 4; ++j) b[j] = load_b_frag(Bt, K, n0 + 16 * j, k, lane);
#pragma unroll
        for (int i = 0; i < 4; ++i)
#pragma unroll
            for (int j = 0; j < 4; ++j) acc[i][j] = wmma16(a[i], b[j], acc[i][j]);
    }

    const int nl = lane & 15, hh = lane >> 4;
#pragma unroll
    for (int i = 0; i < 4; ++i) {
#pragma unroll
        for (int j = 0; j < 4; ++j) {
#pragma unroll
            for (int r = 0; r < 8; ++r) {
                const int m = m0 + 16 * i + r + 8 * hh;   // global row
                const int n = n0 + 16 * j + nl;           // global col
                const float v = acc[i][j][r] + bias[n];
                if (mode == 2) {
                    ((float*)outp)[(size_t)m * N + n] = v;
                } else {
                    const _Float16 hv = (_Float16)v;
                    const int b_ = m >> 11, t = m & (SEQ - 1);
                    const int h_ = n >> 6,  hd = n & (HDIM - 1);
                    if (mode == 0)
                        ((_Float16*)outp)[((size_t)(b_ * HEADS + h_) * SEQ + t) * AUGW + hd] = hv;
                    else
                        ((_Float16*)outp)[((size_t)(b_ * HEADS + h_) * HDIM + hd) * SEQ + t] = hv;
                }
            }
        }
    }
}

// ---------------------------------------------------------------------------
// STP augmentation: one thread per (b,h,t) row.
__global__ void __launch_bounds__(256)
stp_augment(const float* __restrict__ Wqs, const float* __restrict__ bqs,
            const float* __restrict__ Wks, const float* __restrict__ bks,
            const float* __restrict__ Wvs, const float* __restrict__ bvs,
            _Float16* __restrict__ Qa, _Float16* __restrict__ Ka,
            const _Float16* __restrict__ Vt) {
    const int idx = blockIdx.x * 256 + threadIdx.x;
    if (idx >= BHT) return;
    const int bh = idx >> 11;          // b*H + h
    const int t  = idx & (SEQ - 1);

    _Float16* qrow = Qa + (size_t)idx * AUGW;
    _Float16* krow = Ka + (size_t)idx * AUGW;
    const _Float16* vcol = Vt + (size_t)bh * HDIM * SEQ + t;

    float q0 = bqs[0], q1 = bqs[1], q2 = bqs[2];
    float k0 = bks[0], k1 = bks[1], k2 = bks[2];
    float v0 = bvs[0], v1 = bvs[1], v2 = bvs[2];
#pragma unroll 8
    for (int hd = 0; hd < HDIM; ++hd) {
        const float qv = (float)qrow[hd];
        const float kv = (float)krow[hd];
        const float vv = (float)vcol[(size_t)hd * SEQ];
        q0 += Wqs[hd] * qv;  q1 += Wqs[HDIM + hd] * qv;  q2 += Wqs[2 * HDIM + hd] * qv;
        k0 += Wks[hd] * kv;  k1 += Wks[HDIM + hd] * kv;  k2 += Wks[2 * HDIM + hd] * kv;
        v0 += Wvs[hd] * vv;  v1 += Wvs[HDIM + hd] * vv;  v2 += Wvs[2 * HDIM + hd] * vv;
    }
    // cross(k3, v3)
    const float c0 = k1 * v2 - k2 * v1;
    const float c1 = k2 * v0 - k0 * v2;
    const float c2 = k0 * v1 - k1 * v0;
    qrow[64] = (_Float16)q0; qrow[65] = (_Float16)q1; qrow[66] = (_Float16)q2;
    krow[64] = (_Float16)c0; krow[65] = (_Float16)c1; krow[66] = (_Float16)c2;
#pragma unroll
    for (int z = 67; z < AUGW; ++z) { qrow[z] = (_Float16)0.0f; krow[z] = (_Float16)0.0f; }
}

// ---------------------------------------------------------------------------
// Flash attention. One wave (32 threads) per (b,h, 16-query-row tile).
__global__ void __launch_bounds__(32)
flash_attn(const _Float16* __restrict__ Qa, const _Float16* __restrict__ Ka,
           const _Float16* __restrict__ Vt, const float* __restrict__ tempP,
           _Float16* __restrict__ Obuf) {
    __shared__ __align__(16) _Float16 lp[16 * 32];   // P tile relayout bounce

    const int lane = threadIdx.x;
    const int bh   = blockIdx.x >> 7;        // b*H + h   (T/16 = 128 q tiles)
    const int qt   = blockIdx.x & 127;
    const float temp = tempP[0];

    const _Float16* Qrows = Qa + ((size_t)bh * SEQ + qt * 16) * AUGW;
    const _Float16* Krows = Ka + (size_t)bh * SEQ * AUGW;
    const _Float16* Vbase = Vt + (size_t)bh * HDIM * SEQ;

    // Query A-fragments for K = 96 = 3 chunks of 32, resident all loop long.
    v16h aq[3];
#pragma unroll
    for (int c = 0; c < 3; ++c) aq[c] = load_a_frag(Qrows, AUGW, 0, 32 * c, lane);

    v8f o[4];
#pragma unroll
    for (int tn = 0; tn < 4; ++tn) o[tn] = zero_v8f();
    float mrow[8], lrow[8];
#pragma unroll
    for (int r = 0; r < 8; ++r) { mrow[r] = -1e30f; lrow[r] = 0.0f; }

    const int nl = lane & 15, hh = lane >> 4;

#pragma unroll 1
    for (int j = 0; j < SEQ / 32; ++j) {       // 32-key blocks
        const int kb = j * 32;

        // S = Qa @ Ka^T over K=96 (two 16-key tiles)
        v8f s0 = zero_v8f(), s1 = zero_v8f();
#pragma unroll
        for (int c = 0; c < 3; ++c) {
            v16h b0 = load_b_frag(Krows, AUGW, kb,      32 * c, lane);
            v16h b1 = load_b_frag(Krows, AUGW, kb + 16, 32 * c, lane);
            s0 = wmma16(aq[c], b0, s0);
            s1 = wmma16(aq[c], b1, s1);
        }

        // online softmax per row (row r+8*hh lives across the 16-lane half)
#pragma unroll
        for (int r = 0; r < 8; ++r) {
            const float sa = s0[r] * temp;
            const float sb = s1[r] * temp;
            float mx = fmaxf(sa, sb);
#pragma unroll
            for (int off = 8; off >= 1; off >>= 1)
                mx = fmaxf(mx, __shfl_xor(mx, off, 32));
            const float mn    = fmaxf(mrow[r], mx);
            const float alpha = __expf(mrow[r] - mn);
            const float p0    = __expf(sa - mn);
            const float p1    = __expf(sb - mn);
            float rs = p0 + p1;
#pragma unroll
            for (int off = 8; off >= 1; off >>= 1)
                rs += __shfl_xor(rs, off, 32);
            lrow[r] = lrow[r] * alpha + rs;
            mrow[r] = mn;
#pragma unroll
            for (int tn = 0; tn < 4; ++tn) o[tn][r] *= alpha;
            // stash P in [row][key] layout for A-fragment reload
            lp[(r + 8 * hh) * 32 + nl]      = (_Float16)p0;
            lp[(r + 8 * hh) * 32 + 16 + nl] = (_Float16)p1;
        }
        __syncthreads();                        // single-wave WG: cheap
        const v16h ap = load_a_frag(lp, 32, 0, 0, lane);
        __syncthreads();

        // O += P(16x32) @ V(32x64)
#pragma unroll
        for (int tn = 0; tn < 4; ++tn) {
            v16h bv = load_b_frag(Vbase, SEQ, tn * 16, kb, lane);
            o[tn] = wmma16(ap, bv, o[tn]);
        }
    }

    // epilogue: normalize and scatter to [b, t, h*64 + col] (f16)
    const int b_ = bh >> 4, h_ = bh & (HEADS - 1);
#pragma unroll
    for (int tn = 0; tn < 4; ++tn) {
#pragma unroll
        for (int r = 0; r < 8; ++r) {
            const int row  = qt * 16 + r + 8 * hh;
            const int gcol = h_ * HDIM + tn * 16 + nl;
            Obuf[((size_t)b_ * SEQ + row) * DIM + gcol] =
                (_Float16)(o[tn][r] / lrow[r]);
        }
    }
}

// ---------------------------------------------------------------------------
extern "C" void kernel_launch(void* const* d_in, const int* in_sizes, int n_in,
                              void* d_out, int out_size, void* d_ws, size_t ws_size,
                              hipStream_t stream) {
    (void)in_sizes; (void)n_in; (void)out_size;

    const float* x    = (const float*)d_in[0];
    const float* Wq   = (const float*)d_in[1];
    const float* bq   = (const float*)d_in[2];
    const float* Wk   = (const float*)d_in[3];
    const float* bk   = (const float*)d_in[4];
    const float* Wv   = (const float*)d_in[5];
    const float* bv   = (const float*)d_in[6];
    const float* Wo   = (const float*)d_in[7];
    const float* bo   = (const float*)d_in[8];
    const float* Wqs  = (const float*)d_in[9];
    const float* bqs  = (const float*)d_in[10];
    const float* Wks  = (const float*)d_in[11];
    const float* bks  = (const float*)d_in[12];
    const float* Wvs  = (const float*)d_in[13];
    const float* bvs  = (const float*)d_in[14];
    const float* temp = (const float*)d_in[15];

    // workspace carve-up (element counts)
    const size_t nX  = (size_t)MROWS * DIM;          // 4'194'304
    const size_t nW  = (size_t)DIM * DIM;            // 1'048'576
    const size_t nQa = (size_t)BHT * AUGW;           // 6'291'456
    const size_t nV  = (size_t)BHT * HDIM;           // 4'194'304
    _Float16* x16  = (_Float16*)d_ws;
    _Float16* wq16 = x16  + nX;
    _Float16* wk16 = wq16 + nW;
    _Float16* wv16 = wk16 + nW;
    _Float16* wo16 = wv16 + nW;
    _Float16* Qa   = wo16 + nW;
    _Float16* Ka   = Qa + nQa;
    _Float16* Vt   = Ka + nQa;
    _Float16* Obuf = Vt + nV;
    const size_t needBytes = 2 * (nX + 4 * nW + 2 * nQa + nV + nX);
    if (ws_size < needBytes) return;

    // 1) f32 -> f16 converts
    cvt_f32_to_f16<<<(int)((nX + 255) / 256), 256, 0, stream>>>(x,  x16,  (int)nX);
    cvt_f32_to_f16<<<(int)((nW + 255) / 256), 256, 0, stream>>>(Wq, wq16, (int)nW);
    cvt_f32_to_f16<<<(int)((nW + 255) / 256), 256, 0, stream>>>(Wk, wk16, (int)nW);
    cvt_f32_to_f16<<<(int)((nW + 255) / 256), 256, 0, stream>>>(Wv, wv16, (int)nW);
    cvt_f32_to_f16<<<(int)((nW + 255) / 256), 256, 0, stream>>>(Wo, wo16, (int)nW);

    // 2) projections (1024 blocks = 64 x 16 tiles of 64x64)
    const int gemmGrid = (MROWS / 64) * (DIM / 64);
    gemm_wmma_64x64<<<gemmGrid, 32, 0, stream>>>(x16, wq16, bq, Qa, MROWS, DIM, DIM, 0);
    gemm_wmma_64x64<<<gemmGrid, 32, 0, stream>>>(x16, wk16, bk, Ka, MROWS, DIM, DIM, 0);
    gemm_wmma_64x64<<<gemmGrid, 32, 0, stream>>>(x16, wv16, bv, Vt, MROWS, DIM, DIM, 1);

    // 3) STP augmentation (+ zero the 67..95 pad)
    stp_augment<<<BHT / 256, 256, 0, stream>>>(Wqs, bqs, Wks, bks, Wvs, bvs, Qa, Ka, Vt);

    // 4) flash attention: B*H*(T/16) = 4096 waves
    flash_attn<<<BATCH * HEADS * (SEQ / 16), 32, 0, stream>>>(Qa, Ka, Vt, temp, Obuf);

    // 5) output projection, f32 result + bias
    gemm_wmma_64x64<<<gemmGrid, 32, 0, stream>>>(Obuf, wo16, bo, (float*)d_out,
                                                 MROWS, DIM, DIM, 2);
}
